// MyModel_87522843559774
// MI455X (gfx1250) — matrix operand order repaired and checked by reading, not compiled, not run on hardware
//
#include <hip/hip_runtime.h>

#if defined(__has_builtin)
#if __has_builtin(__builtin_amdgcn_global_load_async_to_lds_b128)
#define HAVE_ASYNC_LDS 1
#endif
#endif

typedef __bf16 bf16_t;
typedef bf16_t v16bf __attribute__((ext_vector_type(16)));
typedef float  v8f   __attribute__((ext_vector_type(8)));
typedef int    v4i   __attribute__((ext_vector_type(4)));
typedef __attribute__((address_space(1))) v4i* gv4i_p;   // global b128 pointer
typedef __attribute__((address_space(3))) v4i* lv4i_p;   // LDS b128 pointer

#define T_STEPS 2048
#define BATCH   32
#define FEAT    256
#define UNITSN  512
#define NSLICE  16     // workgroups per direction
#define SLICE   32     // h columns owned per workgroup
#define ZW      128    // gate columns per workgroup (4*SLICE)
#define KCH     24     // (FEAT+UNITSN)/32 k-chunks
#define NWAVE   8

// LDS layout (bytes)
#define LDSB_OFF 0         // 24*8*32*32B  = 196608 (weight B-fragments, bf16)
#define LDSA_OFF 196608    // 24*2*32*32B  = 49152  (A-fragments: [x_t | h], bf16)
#define ZBUF_OFF 245760    // 32*128*4B    = 16384  (z slice, f32)
#define BIAS_OFF 262144    // 128*4B       = 512
#define LDS_TOTAL 262656

// workspace layout
#define HFRAG_U16 16384            // u16 per (buf,dir): 16 sl * 2 m * 32 lane * 16 e
#define HFRAG_BYTES 131072         // 2 bufs * 2 dirs * 16384 * 2B
#define CTR_OFF   131072
#define XFRAG_OFF 262144           // byte offset of packed-x region
#define XFRAG_U16_PER_T 8192       // 8 kc * 2 m * 32 lane * 16 e

__device__ __forceinline__ unsigned short f2bf(float f) {
  unsigned u = __float_as_uint(f);
  unsigned r = (u + 0x7FFFu + ((u >> 16) & 1u)) >> 16;  // round-to-nearest-even
  return (unsigned short)r;
}

// A-fragment (16x32 bf16) index helpers: lanes 0-15 hold K in {0..7,16..23},
// lanes 16-31 hold {8..15,24..31}; element e: K = (e<8?0:16) + khalf*8 + (e&7).
__device__ __forceinline__ int fragLane(int row, int kk) { return row | (((kk >> 3) & 1) << 4); }
__device__ __forceinline__ int fragElem(int kk)          { return (kk & 7) | ((kk >> 4) << 3); }

__device__ __forceinline__ void stageA(unsigned short* ldsA, int b, int k, float v) {
  int kc = k >> 5, kk = k & 31;
  int m = b >> 4, row = b & 15;
  ldsA[(((kc * 2 + m) * 32) + fragLane(row, kk)) * 16 + fragElem(kk)] = f2bf(v);
}

// 16B global -> LDS copy; async-DMA path when the toolchain exposes it.
// Generic LDS pointers carry the LDS offset in the low 32 bits (ISA aperture rule),
// so the AS3 pointer can be formed by integer truncation.
__device__ __forceinline__ void copy16(const void* g, void* l) {
#ifdef HAVE_ASYNC_LDS
  __builtin_amdgcn_global_load_async_to_lds_b128(
      (gv4i_p)(size_t)g, (lv4i_p)(unsigned)(size_t)l, 0, 0);
#else
  *(float4*)l = *(const float4*)g;
#endif
}
__device__ __forceinline__ void wait_async() {
#ifdef HAVE_ASYNC_LDS
#if __has_builtin(__builtin_amdgcn_s_wait_asynccnt)
  __builtin_amdgcn_s_wait_asynccnt(0);
#else
  asm volatile("s_wait_asynccnt 0x0" ::: "memory");
#endif
#endif
}

__device__ __forceinline__ float sigf(float x)  { return 1.0f / (1.0f + __expf(-x)); }
__device__ __forceinline__ float tanhf_(float x){ return 2.0f / (1.0f + __expf(-2.0f * x)) - 1.0f; }

// ---- prologue: pack x[B,T,F] into bf16 A-fragment order per timestep ----
__global__ __launch_bounds__(256) void xpack(const float* __restrict__ x,
                                             unsigned short* __restrict__ Xfrag) {
  const int t = blockIdx.x;
  const int tid = threadIdx.x;
  #pragma unroll 4
  for (int i = 0; i < 32; ++i) {
    int p = tid * 32 + i;
    int b = p >> 8, f = p & 255;
    float v = x[((size_t)b * T_STEPS + t) * FEAT + f];
    int kc = f >> 5, kk = f & 31;
    int m = b >> 4, row = b & 15;
    Xfrag[(size_t)t * XFRAG_U16_PER_T +
          (((kc * 2 + m) * 32) + fragLane(row, kk)) * 16 + fragElem(kk)] = f2bf(v);
  }
}

template <bool XFRAG>
__global__ __launch_bounds__(256) void bilstm_persistent(
    const float* __restrict__ x, const unsigned short* __restrict__ Xfrag,
    const float* __restrict__ Wf, const float* __restrict__ Uf, const float* __restrict__ bfv,
    const float* __restrict__ Wb, const float* __restrict__ Ub, const float* __restrict__ bbv,
    float* __restrict__ out, unsigned short* __restrict__ Hfrag, unsigned* __restrict__ ctrs)
{
  extern __shared__ char smem[];
  unsigned short* ldsB  = (unsigned short*)(smem + LDSB_OFF);
  unsigned short* ldsA  = (unsigned short*)(smem + LDSA_OFF);
  float*          zbuf  = (float*)(smem + ZBUF_OFF);
  float*          biasl = (float*)(smem + BIAS_OFF);

  const int tid  = threadIdx.x;
  const int wv   = tid >> 5;          // wave id 0..7 == N-tile id
  const int lane = tid & 31;
  const int dir  = blockIdx.x >> 4;   // 0 = forward, 1 = backward
  const int sl   = blockIdx.x & 15;
  const int js   = sl * SLICE;

  const float* Wd = dir ? Wb : Wf;
  const float* Ud = dir ? Ub : Uf;
  const float* bd = dir ? bbv : bfv;
  unsigned* ctr = ctrs + dir * 16;    // 64B apart per direction

  // ---- one-time: pack [W;U] column slice into WMMA B-fragment order (bf16) ----
  // B-frag (32x16 KxN): lane l holds N=l&15, K=(e<8?0:16)+(l>>4)*8+(e&7)
  for (int p = tid; p < KCH * NWAVE * 32 * 16; p += 256) {
    int e  = p & 15;
    int l  = (p >> 4) & 31;
    int w  = (p >> 9) & 7;
    int kc = p >> 12;
    int n  = w * 16 + (l & 15);
    int k  = kc * 32 + ((e & 7) + ((e >> 3) << 4) + ((l >> 4) << 3));
    int col = (n >> 5) * 512 + js + (n & 31);
    float v = (k < FEAT) ? Wd[k * 2048 + col] : Ud[(k - FEAT) * 2048 + col];
    ldsB[p] = f2bf(v);
  }
  if (tid < ZW) biasl[tid] = bd[(tid >> 5) * 512 + js + (tid & 31)];
  __syncthreads();

  // ---- preload x-fragments for t = 0 ----
  if (XFRAG) {
    const char* gx = (const char*)(Xfrag + (size_t)(dir ? T_STEPS - 1 : 0) * XFRAG_U16_PER_T);
    #pragma unroll
    for (int i = 0; i < 4; ++i) { int off = (tid + i * 256) * 16; copy16(gx + off, (char*)ldsA + off); }
  } else {
    int p0 = tid * 32, b = p0 >> 8, f0 = p0 & 255;
    const float* src = x + ((size_t)b * T_STEPS + (dir ? T_STEPS - 1 : 0)) * FEAT + f0;
    #pragma unroll 8
    for (int i = 0; i < 32; ++i) stageA(ldsA, b, f0 + i, src[i]);
  }

  float creg[4] = {0.f, 0.f, 0.f, 0.f};

  for (int t = 0; t < T_STEPS; ++t) {
    const int tt = dir ? (T_STEPS - 1 - t) : t;

    // ---- (1) linear copy of h fragments (32KB) into A-frag chunks 8..23 ----
    {
      const char* gh = (const char*)(Hfrag + ((size_t)((t & 1) * 2 + dir)) * HFRAG_U16);
      char* ld = (char*)ldsA + 16384;
      #pragma unroll
      for (int i = 0; i < 8; ++i) { int off = (tid + i * 256) * 16; copy16(gh + off, ld + off); }
    }
    wait_async();            // also covers the x-frag copy issued last step
    __syncthreads();

    // ---- (2) z-slice[32x128] = [x_t | h] @ [W;U] slice, pipelined bf16 WMMA ----
    v8f acc0 = {}, acc1 = {};
    {
      const unsigned short* Bb = ldsB + (wv * 32 + lane) * 16;  // +4096 u16 per kc
      const unsigned short* Ab = ldsA + lane * 16;              // +1024 per kc, +512 per m
      v16bf cb  = *(const v16bf*)(Bb);
      v16bf ca0 = *(const v16bf*)(Ab);
      v16bf ca1 = *(const v16bf*)(Ab + 512);
      #pragma unroll
      for (int kc = 0; kc < KCH; ++kc) {
        int kn = (kc + 1 < KCH) ? kc + 1 : kc;
        v16bf nb  = *(const v16bf*)(Bb + kn * 4096);
        v16bf na0 = *(const v16bf*)(Ab + kn * 1024);
        v16bf na1 = *(const v16bf*)(Ab + kn * 1024 + 512);
        acc0 = __builtin_amdgcn_wmma_f32_16x16x32_bf16(false, ca0, false, cb,
                                                       (short)0, acc0, false, false);
        acc1 = __builtin_amdgcn_wmma_f32_16x16x32_bf16(false, ca1, false, cb,
                                                       (short)0, acc1, false, false);
        cb = nb; ca0 = na0; ca1 = na1;
      }
    }
    // C/D layout: VGPR r -> M = r + (lane>>4)*8, N = lane&15
    {
      int n    = wv * 16 + (lane & 15);
      int half = (lane >> 4) * 8;
      #pragma unroll
      for (int r = 0; r < 8; ++r) {
        zbuf[(half + r) * ZW + n]      = acc0[r];
        zbuf[(16 + half + r) * ZW + n] = acc1[r];
      }
    }
    __syncthreads();

    // ---- (4) start x-fragment copy for t+1 (overlaps gate math + barrier) ----
    if (t + 1 < T_STEPS) {
      if (XFRAG) {
        int tn = dir ? tt - 1 : tt + 1;
        const char* gx = (const char*)(Xfrag + (size_t)tn * XFRAG_U16_PER_T);
        #pragma unroll
        for (int i = 0; i < 4; ++i) { int off = (tid + i * 256) * 16; copy16(gx + off, (char*)ldsA + off); }
        if (t + 2 < T_STEPS) {
          int tn2 = dir ? tt - 2 : tt + 2;
          __builtin_prefetch(Xfrag + (size_t)tn2 * XFRAG_U16_PER_T + tid * 32, 0, 1);
        }
      } else {
        int tn = dir ? tt - 1 : tt + 1;
        int p0 = tid * 32, b = p0 >> 8, f0 = p0 & 255;
        const float* src = x + ((size_t)b * T_STEPS + tn) * FEAT + f0;
        if (t + 2 < T_STEPS)
          __builtin_prefetch(x + ((size_t)b * T_STEPS + (dir ? tn - 1 : tn + 1)) * FEAT + f0, 0, 1);
        #pragma unroll 8
        for (int i = 0; i < 32; ++i) stageA(ldsA, b, f0 + i, src[i]);
      }
    }

    // ---- (5) gates + state update; write h directly as bf16 fragments ----
    unsigned short* HW = Hfrag + ((size_t)(((t & 1) ^ 1) * 2 + dir)) * HFRAG_U16;
    #pragma unroll
    for (int q = 0; q < 4; ++q) {
      int p  = tid * 4 + q;
      int b  = p >> 5, jj = p & 31;
      float zi = zbuf[b * ZW + jj]      + biasl[jj];
      float zf = zbuf[b * ZW + 32 + jj] + biasl[32 + jj];
      float zg = zbuf[b * ZW + 64 + jj] + biasl[64 + jj];
      float zo = zbuf[b * ZW + 96 + jj] + biasl[96 + jj];
      float c  = sigf(zf) * creg[q] + sigf(zi) * tanhf_(zg);
      float h  = sigf(zo) * tanhf_(c);
      creg[q] = c;
      // this WG's h slice is exactly A-frag chunk kc = 8+sl (k = 256+js+jj, jj<32)
      HW[((sl * 2 + (b >> 4)) * 32 + fragLane(b & 15, jj)) * 16 + fragElem(jj)] = f2bf(h);
      if (t == T_STEPS - 1) {
        int j = js + jj;
        out[b * 1024 + dir * 512 + j]               = h;  // merged [32,1024]
        out[(dir ? 65536 : 32768) + b * UNITSN + j] = h;  // h_f / h_b
        out[(dir ? 81920 : 49152) + b * UNITSN + j] = c;  // c_f / c_b
      }
    }

    // ---- (6) release h, sync the 16 WGs of this direction, acquire ----
    __threadfence();
    __syncthreads();
    __builtin_amdgcn_s_cluster_barrier();   // NOP unless cluster-dispatched
    if (tid == 0) {
      unsigned tgt = (unsigned)(NSLICE * (t + 1));
      __hip_atomic_fetch_add(ctr, 1u, __ATOMIC_RELEASE, __HIP_MEMORY_SCOPE_AGENT);
      while (__hip_atomic_load(ctr, __ATOMIC_ACQUIRE, __HIP_MEMORY_SCOPE_AGENT) < tgt)
        __builtin_amdgcn_s_sleep(2);
    }
    __syncthreads();
    __builtin_amdgcn_fence(__ATOMIC_ACQUIRE, "agent");  // every wave invalidates its L0
  }
}

extern "C" void kernel_launch(void* const* d_in, const int* in_sizes, int n_in,
                              void* d_out, int out_size, void* d_ws, size_t ws_size,
                              hipStream_t stream) {
  (void)in_sizes; (void)n_in; (void)out_size;
  const float* x   = (const float*)d_in[0];
  const float* Wf  = (const float*)d_in[1];
  const float* Uf  = (const float*)d_in[2];
  const float* bfv = (const float*)d_in[3];
  const float* Wb  = (const float*)d_in[4];
  const float* Ub  = (const float*)d_in[5];
  const float* bbv = (const float*)d_in[6];
  float* out = (float*)d_out;

  unsigned short* Hfrag = (unsigned short*)d_ws;
  unsigned*       ctrs  = (unsigned*)((char*)d_ws + CTR_OFF);
  unsigned short* Xfrag = (unsigned short*)((char*)d_ws + XFRAG_OFF);
  (void)hipMemsetAsync(d_ws, 0, HFRAG_BYTES + 256, stream);

  const size_t xfrag_need = (size_t)XFRAG_OFF + (size_t)T_STEPS * XFRAG_U16_PER_T * 2;
  dim3 grid(2 * NSLICE), block(256);
  if (ws_size >= xfrag_need) {
    xpack<<<T_STEPS, 256, 0, stream>>>(x, Xfrag);
    bilstm_persistent<true><<<grid, block, LDS_TOTAL, stream>>>(
        x, Xfrag, Wf, Uf, bfv, Wb, Ub, bbv, out, Hfrag, ctrs);
  } else {
    bilstm_persistent<false><<<grid, block, LDS_TOTAL, stream>>>(
        x, Xfrag, Wf, Uf, bfv, Wb, Ub, bbv, out, Hfrag, ctrs);
  }
}
